// PAM_61649960566930
// MI455X (gfx1250) — matrix operand order repaired and checked by reading, not compiled, not run on hardware
//
#include <hip/hip_runtime.h>

typedef _Float16 f16;
typedef __attribute__((ext_vector_type(16))) _Float16     v16h;
typedef __attribute__((ext_vector_type(2)))  _Float16     v2h;
typedef __attribute__((ext_vector_type(8)))  float        v8f;
typedef __attribute__((ext_vector_type(8)))  unsigned int v8u;
typedef unsigned int   u32;
typedef unsigned short u16;

#define CCH   256
#define CIQ   32
#define NPOS  4096
#define JT    (NPOS / 32)
#define L2E   1.44269504088896f

// ---------------------------------------------------------------------------
// Projection kernel: Q = (Wc*x+bc)*log2(e)  [folded for base-2 softmax],
// K = Wb*x+bb, V = Wd*x+bd, stored f16 as Q[b][n][32], K[b][n][32],
// V[b][n][256] (position-major for flash attention).
// grid = B * (N/32) = 1024 blocks, 256 threads.
// ---------------------------------------------------------------------------
__global__ __launch_bounds__(256) void pam_proj(
    const float* __restrict__ x,
    const float* __restrict__ Wb, const float* __restrict__ bb,
    const float* __restrict__ Wc, const float* __restrict__ bc,
    const float* __restrict__ Wd, const float* __restrict__ bd,
    f16* __restrict__ Qo, f16* __restrict__ Ko, f16* __restrict__ Vo)
{
    __shared__ float xs[CCH * 33];          // x tile [c][n], padded rows (33.8 KB)
    const int t  = threadIdx.x;
    const int b  = blockIdx.x >> 7;
    const int n0 = (blockIdx.x & 127) * 32;

    // coalesced load of x[b][:, n0:n0+32]
    const float* xb = x + ((size_t)b * CCH) * NPOS + n0;
    #pragma unroll 4
    for (int i = 0; i < 32; ++i) {
        int c = (t >> 5) + i * 8;
        int n = t & 31;
        xs[c * 33 + n] = xb[(size_t)c * NPOS + n];
    }
    __syncthreads();

    const int n = t & 31;   // position within tile (lane)
    const int g = t >> 5;   // wave id: output-channel group
    float accV[32], accQ[4], accK[4];
    #pragma unroll
    for (int j = 0; j < 32; ++j) accV[j] = 0.f;
    #pragma unroll
    for (int j = 0; j < 4; ++j) { accQ[j] = 0.f; accK[j] = 0.f; }

    for (int c = 0; c < CCH; ++c) {
        float xv = xs[c * 33 + n];
        #pragma unroll
        for (int j = 0; j < 4; ++j) {
            accQ[j] = fmaf(Wc[(g * 4 + j) * CCH + c], xv, accQ[j]);
            accK[j] = fmaf(Wb[(g * 4 + j) * CCH + c], xv, accK[j]);
        }
        #pragma unroll
        for (int j = 0; j < 32; ++j)
            accV[j] = fmaf(Wd[(g * 32 + j) * CCH + c], xv, accV[j]);
    }
    __syncthreads();

    // re-stage as f16 in LDS (reuse xs), then coalesced contiguous stores
    f16* st  = (f16*)xs;
    f16* stV = st;                 // 32n x 256c = 8192 f16
    f16* stQ = st + 32 * 256;      // 32n x 32k  = 1024 f16
    f16* stK = stQ + 32 * 32;      // 32n x 32k  = 1024 f16
    #pragma unroll
    for (int j = 0; j < 32; ++j)
        stV[n * 256 + g * 32 + j] = (f16)(accV[j] + bd[g * 32 + j]);
    #pragma unroll
    for (int j = 0; j < 4; ++j) {
        stQ[n * 32 + g * 4 + j] = (f16)((accQ[j] + bc[g * 4 + j]) * L2E);
        stK[n * 32 + g * 4 + j] = (f16)(accK[j] + bb[g * 4 + j]);
    }
    __syncthreads();

    const u32* s32 = (const u32*)xs;
    u32* vg = (u32*)(Vo + ((size_t)b * NPOS + n0) * CCH);
    u32* qg = (u32*)(Qo + ((size_t)b * NPOS + n0) * CIQ);
    u32* kg = (u32*)(Ko + ((size_t)b * NPOS + n0) * CIQ);
    #pragma unroll
    for (int i = 0; i < 16; ++i) vg[t + 256 * i] = s32[t + 256 * i];
    qg[t]       = s32[4096 + t];
    qg[t + 256] = s32[4096 + 256 + t];
    kg[t]       = s32[4608 + t];
    kg[t + 256] = s32[4608 + 256 + t];
}

// ---------------------------------------------------------------------------
// Fused flash-attention with WMMA f16 (f32 accum). Computes S^T = K_tile x Q^T
// (S pre-scaled by log2(e) via Q) so softmax is pure base-2: v_sub + v_exp.
// Reduction over keys is in-lane (+1 xor16 shuffle); per-query stats are
// scalar per lane; K tiles need no LDS transpose. K and V double-buffered via
// async global->LDS DMA (ASYNCcnt). Packed f16 P conversion; LDS-broadcast
// alpha transposition.
// Block = 256 threads (8 waves). Wave w: q-tile = w/2 (16 rows), c-half = w&1
// (128 of 256 V channels). 64 query rows / block; grid = 8 * 64 = 512.
// ---------------------------------------------------------------------------
__global__ __launch_bounds__(256) void pam_attn(
    const f16* __restrict__ Q, const f16* __restrict__ K,
    const f16* __restrict__ V, const float* __restrict__ x,
    const float* __restrict__ gamma_p, float* __restrict__ out)
{
    __shared__ u32 kls32[2][32 * 18];    // K tile [j][k], rows padded to 18 dwords
    __shared__ u32 vs32[2][32 * 129];    // V tile [j][c], 129-dword rows
    __shared__ u32 ps32[8][16 * 17];     // per-wave P / O-transpose scratch
    __shared__ __align__(16) float als[8][16];   // per-wave alpha broadcast

    const int t    = threadIdx.x;
    const int lane = t & 31;
    const int w    = t >> 5;
    const int qt   = w >> 1;             // which 16-row q tile
    const int ch   = w & 1;              // which 128-channel half of V
    const int b    = blockIdx.x >> 6;
    const int n0   = (blockIdx.x & 63) * 64;
    const int m    = lane & 15;          // A-row / D-column index
    const int h    = lane >> 4;          // lane half

    const f16* Kb   = K + ((size_t)b * NPOS) * CIQ;
    const u32* Vb32 = (const u32*)(V + ((size_t)b * NPOS) * CCH);
    u32* psd = ps32[w];

    // ---- Q^T in WMMA B-layout (lane%16 = query, VGPR v = k pair, table order)
    const u16* Qu = (const u16*)(Q + ((size_t)b * NPOS + n0 + qt * 16) * CIQ);
    v8u qb_u;
    #pragma unroll
    for (int v = 0; v < 8; ++v) {
        u32 lo = Qu[m * CIQ + 8 * h + 2 * v + (v >> 2) * 8];
        u32 hi = Qu[m * CIQ + 8 * h + 2 * v + (v >> 2) * 8 + 1];
        qb_u[v] = lo | (hi << 16);
    }
    const v16h qb = __builtin_bit_cast(v16h, qb_u);

    const v8f vzero = {0.f, 0.f, 0.f, 0.f, 0.f, 0.f, 0.f, 0.f};
    v8f   O[8];
    float mold = -1e30f, lsum = 0.f;
    #pragma unroll
    for (int r = 0; r < 8; ++r) O[r] = vzero;

    // ---- async DMA helpers (ASYNCcnt) --------------------------------------
    auto issueK = [&](int buf, int j0k) {       // 2 KB tile, rows padded 16->18
        const u32* ksrc = (const u32*)(Kb + (size_t)j0k * CIQ);
        #pragma unroll
        for (int i = 0; i < 2; ++i) {
            int dw = t + 256 * i;
            u32 lb = (u32)(size_t)&kls32[buf][(dw >> 4) * 18 + (dw & 15)];
            asm volatile("global_load_async_to_lds_b32 %0, %1, %2"
                         :: "v"(lb), "v"(dw * 4), "s"(ksrc) : "memory");
        }
    };
    auto issueV = [&](int buf, int j0v) {       // 16 KB tile, rows padded to 129
        const u32* vsrc = Vb32 + (size_t)j0v * 128;
        #pragma unroll
        for (int i = 0; i < 16; ++i) {
            int dw = t + 256 * i;
            u32 lb = (u32)(size_t)&vs32[buf][(dw >> 7) * 129 + (dw & 127)];
            asm volatile("global_load_async_to_lds_b32 %0, %1, %2"
                         :: "v"(lb), "v"(dw * 4), "s"(vsrc) : "memory");
        }
    };
    // A-layout gather of one 16-key x 32-dim K tile from padded LDS rows
    auto loadKA = [&](const u32* kl, int tile) {
        const u32* row = kl + (tile * 16 + m) * 18;
        uint2 a0 = *(const uint2*)(row + h * 4);
        uint2 a1 = *(const uint2*)(row + h * 4 + 2);
        uint2 b0 = *(const uint2*)(row + 8 + h * 4);
        uint2 b1 = *(const uint2*)(row + 8 + h * 4 + 2);
        v8u u;
        u[0] = a0.x; u[1] = a0.y; u[2] = a1.x; u[3] = a1.y;
        u[4] = b0.x; u[5] = b0.y; u[6] = b1.x; u[7] = b1.y;
        return __builtin_bit_cast(v16h, u);
    };

    // ---- prologue: tile 0 into buffer 0 ------------------------------------
    issueK(0, 0);
    issueV(0, 0);
    asm volatile("s_wait_asynccnt 0x0" ::: "memory");
    __syncthreads();

    for (int jt = 0; jt < JT; ++jt) {
        const int cur = jt & 1;
        const int nxt = cur ^ 1;
        const bool haveNext = (jt + 1 < JT);

        if (haveNext) {                      // stage next tile via async DMA
            issueK(nxt, (jt + 1) * 32);
            issueV(nxt, (jt + 1) * 32);
        }

        // ---- S^T = K_tile x Q^T : D rows = keys (r+8h), cols = queries (m)
        //      (S already scaled by log2(e) through Q)
        v16h ka0 = loadKA(kls32[cur], 0);
        v16h ka1 = loadKA(kls32[cur], 1);
        v8f s0 = __builtin_amdgcn_wmma_f32_16x16x32_f16(false, ka0, false, qb,
                                                        (short)0, vzero, false, false);
        v8f s1 = __builtin_amdgcn_wmma_f32_16x16x32_f16(false, ka1, false, qb,
                                                        (short)0, vzero, false, false);

        // ---- base-2 online softmax: in-lane max tree + single xor16 shuffle
        float mt = fmaxf(s0[0], s1[0]);
        #pragma unroll
        for (int r = 1; r < 8; ++r) mt = fmaxf(mt, fmaxf(s0[r], s1[r]));
        mt = fmaxf(mt, __shfl_xor(mt, 16, 32));

        float mn = fmaxf(mold, mt);
        float al = __builtin_amdgcn_exp2f(mold - mn);   // raw v_exp_f32
        mold = mn;

        float p0[8], p1[8], sum = 0.f;
        #pragma unroll
        for (int r = 0; r < 8; ++r) {
            p0[r] = __builtin_amdgcn_exp2f(s0[r] - mn);
            p1[r] = __builtin_amdgcn_exp2f(s1[r] - mn);
            sum += p0[r] + p1[r];
        }
        sum += __shfl_xor(sum, 16, 32);
        lsum = lsum * al + sum;

        // ---- alpha lane(query) -> row-indexed: 1 LDS store + 2 b128 broadcasts
        als[w][m] = al;                       // halves write identical values
        float4 aa0 = *(const float4*)&als[w][8 * h];
        float4 aa1 = *(const float4*)&als[w][8 * h + 4];
        float arr[8] = {aa0.x, aa0.y, aa0.z, aa0.w, aa1.x, aa1.y, aa1.z, aa1.w};
        #pragma unroll
        for (int cc = 0; cc < 8; ++cc) {
            #pragma unroll
            for (int r = 0; r < 8; ++r) O[cc][r] *= arr[r];
        }

        // ---- P^T (D-layout) -> per-wave LDS [query][key] -> A-layout gather
        //      adjacent keys pack into one dword via v_cvt_pkrtz
        #pragma unroll
        for (int r = 0; r < 8; r += 2) {
            int kd = (r >> 1) + 4 * h;       // dword index = (r + 8h)/2
            u32 q0 = __builtin_bit_cast(u32, __builtin_amdgcn_cvt_pkrtz(p0[r], p0[r + 1]));
            u32 q1 = __builtin_bit_cast(u32, __builtin_amdgcn_cvt_pkrtz(p1[r], p1[r + 1]));
            psd[m * 17 + kd]     = q0;
            psd[m * 17 + 8 + kd] = q1;
        }
        v8u pu;
        #pragma unroll
        for (int v = 0; v < 8; ++v)
            pu[v] = psd[m * 17 + ((v >> 2) * 8 + h * 4 + (v & 3))];
        v16h pa = __builtin_bit_cast(v16h, pu);

        // ---- O += P * V  (8 chunks of 16 channels for this wave's half)
        const u32* v32 = vs32[cur];
        #pragma unroll
        for (int cc = 0; cc < 8; ++cc) {
            v8u vu;
            #pragma unroll
            for (int v = 0; v < 8; ++v) {
                const u16* v16p = (const u16*)v32;
                int k0 = 8 * h + 2 * v + (v >> 2) * 8;
                u32 lo = v16p[k0 * 258 + ch * 128 + cc * 16 + m];
                u32 hi = v16p[(k0 + 1) * 258 + ch * 128 + cc * 16 + m];
                vu[v] = lo | (hi << 16);
            }
            v16h vbm = __builtin_bit_cast(v16h, vu);
            O[cc] = __builtin_amdgcn_wmma_f32_16x16x32_f16(false, pa, false, vbm,
                                                           (short)0, O[cc], false, false);
        }

        if (haveNext)
            asm volatile("s_wait_asynccnt 0x0" ::: "memory");
        __syncthreads();
    }

    // ---- epilogue: normalize, transpose per 16x16 chunk via LDS, fused
    //      out[b][c][n] = gamma * attn_out + x[b][c][n] (coalesced along n)
    const float gmm = gamma_p[0];
    als[w][m] = 1.0f / lsum;
    float4 ii0 = *(const float4*)&als[w][8 * h];
    float4 ii1 = *(const float4*)&als[w][8 * h + 4];
    float ilr[8] = {ii0.x, ii0.y, ii0.z, ii0.w, ii1.x, ii1.y, ii1.z, ii1.w};
    float* otf = (float*)ps32[w];

    for (int cc = 0; cc < 8; ++cc) {
        #pragma unroll
        for (int r = 0; r < 8; ++r)
            otf[m * 17 + (r + 8 * h)] = O[cc][r] * ilr[r];     // [c_local][qrow]
        #pragma unroll
        for (int p = 0; p < 8; ++p) {
            int c_loc = 2 * p + h;
            int cg    = ch * 128 + cc * 16 + c_loc;
            float val = otf[c_loc * 17 + m];
            size_t gi = ((size_t)b * CCH + cg) * NPOS + (size_t)(n0 + qt * 16 + m);
            out[gi] = gmm * val + x[gi];
        }
    }
}

// ---------------------------------------------------------------------------
extern "C" void kernel_launch(void* const* d_in, const int* in_sizes, int n_in,
                              void* d_out, int out_size, void* d_ws, size_t ws_size,
                              hipStream_t stream)
{
    (void)in_sizes; (void)n_in; (void)out_size; (void)ws_size;
    const float* x     = (const float*)d_in[0];
    const float* Wb    = (const float*)d_in[1];
    const float* bb    = (const float*)d_in[2];
    const float* Wc    = (const float*)d_in[3];
    const float* bc    = (const float*)d_in[4];
    const float* Wd    = (const float*)d_in[5];
    const float* bd    = (const float*)d_in[6];
    const float* gamma = (const float*)d_in[7];
    float* out = (float*)d_out;

    char* ws = (char*)d_ws;                     // 20 MB used
    f16* Qw = (f16*)(ws);                       // 8*4096*32  f16 = 2 MB
    f16* Kw = (f16*)(ws + (size_t)(2 << 20));   // 2 MB
    f16* Vw = (f16*)(ws + (size_t)(4 << 20));   // 8*4096*256 f16 = 16 MB

    hipLaunchKernelGGL(pam_proj, dim3(1024), dim3(256), 0, stream,
                       x, Wb, bb, Wc, bc, Wd, bd, Qw, Kw, Vw);
    hipLaunchKernelGGL(pam_attn, dim3(512), dim3(256), 0, stream,
                       Qw, Kw, Vw, x, gamma, out);
}
